// GraphModel_4277787426824
// MI455X (gfx1250) — matrix operand order repaired and checked by reading, not compiled, not run on hardware
//
#include <hip/hip_runtime.h>

// ---------------------------------------------------------------------------
// EdgeConv GNN for MI455X (gfx1250): bf16 WMMA per-edge MLP + f32 atomic scatter
// ---------------------------------------------------------------------------

typedef __attribute__((ext_vector_type(16))) __bf16 v16bf;
typedef __attribute__((ext_vector_type(8)))  __bf16 v8bf;
typedef __attribute__((ext_vector_type(4)))  __bf16 v4bf;
typedef __attribute__((ext_vector_type(8)))  float  v8f;
typedef __attribute__((ext_vector_type(4)))  float  v4f;

constexpr int N_NODES = 50000;
constexpr int N_EDGES = 600000;
constexpr int GNN     = 128;
constexpr int NT      = 8;          // 8 N-tiles of 16 -> 128 cols
constexpr int WAVES   = 4;          // waves per block
constexpr int EPB     = WAVES * 16; // 64 edges per block (600000/64 = 9375 exact)
constexpr int SACT    = GNN + 8;    // LDS row stride for activations (bf16)

// gfx1250 has a hardware v_tanh_f32 transcendental; use it if declared.
__device__ __forceinline__ float fast_tanh(float x) {
#if __has_builtin(__builtin_amdgcn_tanhf)
    return __builtin_amdgcn_tanhf(x);
#else
    return tanhf(x);
#endif
}

// ---- A-fragment loader: 16x32 bf16 A tile, documented gfx1250 layout -------
// lane<16:  M=lane,    K = {e, e+8}      (e<8 -> K=e,      e>=8 -> K=e+8)
// lane>=16: M=lane-16, K = {e+8, e+16}
// => two contiguous 8-element (16B) LDS reads at baseK and baseK+16.
__device__ __forceinline__ v16bf load_afrag(const __bf16* p) {
    v8bf lo = *(const v8bf*)p;
    v8bf hi = *(const v8bf*)(p + 16);
    v16bf a;
#pragma unroll
    for (int j = 0; j < 8; ++j) { a[j] = lo[j]; a[j + 8] = hi[j]; }
    return a;
}

// ---- shared MLP body: 16 edges x (KS1*32 -> 128 -> 128), then scatter ------
template<int KS1, int S1>
__device__ __forceinline__ void mlp16_scatter(
    const __bf16* __restrict__ A1,     // 16 rows, stride S1 (bf16), in LDS
    __bf16*       __restrict__ Act,    // 16 rows, stride SACT (bf16), in LDS
    const int*    __restrict__ dstIdx, // 16 dst node ids, in LDS
    const __bf16* __restrict__ w1p, const float* __restrict__ b1,
    const __bf16* __restrict__ w2p, const float* __restrict__ b2,
    float* __restrict__ out, int lane)
{
    const int r     = lane & 15;
    const int hf    = lane >> 4;
    const int baseK = hf * 8;

    // ---------------- GEMM1: [16 x 32*KS1] x [32*KS1 x 128], C = bias ------
    v8f acc[NT];
#pragma unroll
    for (int nt = 0; nt < NT; ++nt) {
        float bv = b1[nt * 16 + r];
#pragma unroll
        for (int j = 0; j < 8; ++j) acc[nt][j] = bv;
    }
#pragma unroll
    for (int kt = 0; kt < KS1; ++kt) {
        v16bf af = load_afrag(A1 + r * S1 + kt * 32 + baseK);
        const __bf16* wp = w1p + ((long)kt * NT) * 512 + lane * 16;
#pragma unroll
        for (int nt = 0; nt < NT; ++nt) {
            v16bf bf = *(const v16bf*)(wp + nt * 512);
            acc[nt] = __builtin_amdgcn_wmma_f32_16x16x32_bf16(
                false, af, false, bf, (short)0, acc[nt], false, false);
        }
    }

    // ---------------- tanh (hw v_tanh_f32) + repack to LDS (bf16) ----------
    // C/D layout: element (vgpr rr, lane) -> row M = rr + 8*hf, col = nt*16 + r
#pragma unroll
    for (int nt = 0; nt < NT; ++nt) {
#pragma unroll
        for (int rr = 0; rr < 8; ++rr) {
            float t = fast_tanh(acc[nt][rr]);
            Act[(rr + hf * 8) * SACT + nt * 16 + r] = (__bf16)t;
        }
    }
    __syncthreads();

    // ---------------- GEMM2: [16 x 128] x [128 x 128], C = bias ------------
    v8f acc2[NT];
#pragma unroll
    for (int nt = 0; nt < NT; ++nt) {
        float bv = b2[nt * 16 + r];
#pragma unroll
        for (int j = 0; j < 8; ++j) acc2[nt][j] = bv;
    }
#pragma unroll
    for (int kt = 0; kt < 4; ++kt) {
        v16bf af = load_afrag(Act + r * SACT + kt * 32 + baseK);
        const __bf16* wp = w2p + ((long)kt * NT) * 512 + lane * 16;
#pragma unroll
        for (int nt = 0; nt < NT; ++nt) {
            v16bf bf = *(const v16bf*)(wp + nt * 512);
            acc2[nt] = __builtin_amdgcn_wmma_f32_16x16x32_bf16(
                false, af, false, bf, (short)0, acc2[nt], false, false);
        }
    }

    // ---------------- scatter-add (segment_sum over dst) -------------------
#pragma unroll
    for (int nt = 0; nt < NT; ++nt) {
        int col = nt * 16 + r;
#pragma unroll
        for (int rr = 0; rr < 8; ++rr) {
            int node = dstIdx[rr + hf * 8];
            atomicAdd(out + (long)node * GNN + col, acc2[nt][rr]);
        }
    }
}

// ---- layer 0: in_dim = 7, 2d = 14 zero-padded to K = 32 --------------------
__global__ __launch_bounds__(WAVES * 32)
void edgeconv_first(const float* __restrict__ x,
                    const int* __restrict__ src, const int* __restrict__ dst,
                    const __bf16* __restrict__ w1p, const float* __restrict__ b1,
                    const __bf16* __restrict__ w2p, const float* __restrict__ b2,
                    float* __restrict__ out)
{
    constexpr int S1 = 40; // 32 + 8 pad (bf16), 80B row stride (16B aligned)
    __shared__ alignas(16) __bf16 ldsA[WAVES][16][S1];
    __shared__ alignas(16) __bf16 ldsAct[WAVES][16][SACT];
    __shared__ int ldsDst[WAVES][16];

    const int lane = threadIdx.x & 31;
    const int wv   = threadIdx.x >> 5;
    const int r    = lane & 15;
    const int hf   = lane >> 4;
    const int e    = blockIdx.x * EPB + wv * 16 + r;

    {
        int si = src[e], di = dst[e];
        if (hf == 0) ldsDst[wv][r] = di;
        const float* xs = x + (long)si * 7;
        const float* xd = x + (long)di * 7;
#pragma unroll
        for (int j = 0; j < 16; ++j) {
            int c = hf * 16 + j;
            float v = 0.f;
            if (c < 7)       v = xd[c];
            else if (c < 14) v = xs[c - 7] - xd[c - 7];
            ldsA[wv][r][c] = (__bf16)v;
        }
    }
    __syncthreads();

    mlp16_scatter<1, S1>(&ldsA[wv][0][0], &ldsAct[wv][0][0], &ldsDst[wv][0],
                         w1p, b1, w2p, b2, out, lane);
}

// ---- layers 1..3: d = 128, K1 = 256 ----------------------------------------
__global__ __launch_bounds__(WAVES * 32)
void edgeconv_mid(const float* __restrict__ h,
                  const int* __restrict__ src, const int* __restrict__ dst,
                  const __bf16* __restrict__ w1p, const float* __restrict__ b1,
                  const __bf16* __restrict__ w2p, const float* __restrict__ b2,
                  float* __restrict__ out)
{
    constexpr int S1 = 264; // 256 + 8 pad (bf16), 528B row stride
    __shared__ alignas(16) __bf16 ldsA[WAVES][16][S1];
    __shared__ alignas(16) __bf16 ldsAct[WAVES][16][SACT];
    __shared__ int ldsDst[WAVES][16];

    const int lane = threadIdx.x & 31;
    const int wv   = threadIdx.x >> 5;
    const int r    = lane & 15;
    const int hf   = lane >> 4;
    const int e    = blockIdx.x * EPB + wv * 16 + r;

    {
        int si = src[e], di = dst[e];
        if (hf == 0) ldsDst[wv][r] = di;
        const float* hs = h + (long)si * GNN + hf * 64;
        const float* hd = h + (long)di * GNN + hf * 64;
#pragma unroll
        for (int c0 = 0; c0 < 64; c0 += 4) {
            v4f a = *(const v4f*)(hd + c0);
            v4f b = *(const v4f*)(hs + c0);
            int col = hf * 64 + c0;
            v4bf pa, pd;
#pragma unroll
            for (int j = 0; j < 4; ++j) {
                pa[j] = (__bf16)a[j];
                pd[j] = (__bf16)(b[j] - a[j]);
            }
            *(v4bf*)&ldsA[wv][r][col]       = pa;  // one ds_store_b64
            *(v4bf*)&ldsA[wv][r][GNN + col] = pd;  // one ds_store_b64
        }
    }
    __syncthreads();

    mlp16_scatter<8, S1>(&ldsA[wv][0][0], &ldsAct[wv][0][0], &ldsDst[wv][0],
                         w1p, b1, w2p, b2, out, lane);
}

// ---- weight pre-swizzle into fragment-ready bf16 layout --------------------
// B tile (kt,nt): 32 lanes x 16 contiguous bf16. K=kt*32+8*(lane/16)+(e&7)+16*(e>>3)
// (matches A); N = nt*16 + lane%16. One 32B load per fragment at runtime.
__global__ void prep_weight(const float* __restrict__ W, int Kreal, int Kpad,
                            __bf16* __restrict__ outp)
{
    int flat = blockIdx.x * blockDim.x + threadIdx.x;
    if (flat >= Kpad * GNN) return;
    int e    = flat & 15;
    int ln   = (flat >> 4) & 31;
    int nt   = (flat >> 9) & 7;
    int kt   = flat >> 12;
    int k    = kt * 32 + (ln >> 4) * 8 + (e & 7) + (e >> 3) * 16;
    int n    = nt * 16 + (ln & 15);
    float v  = (k < Kreal) ? W[(long)k * GNN + n] : 0.f;
    outp[flat] = (__bf16)v;
}

__global__ void zero_kernel(float* __restrict__ p, long n) {
    long i = (long)blockIdx.x * blockDim.x + threadIdx.x;
    if (i < n) p[i] = 0.f;
}

// ---- output head: [N,128] x [128,3] + bias ---------------------------------
__global__ void head_kernel(const float* __restrict__ h,
                            const float* __restrict__ Wo,
                            const float* __restrict__ bo,
                            float* __restrict__ out, int n)
{
    int i = blockIdx.x * blockDim.x + threadIdx.x;
    if (i >= n) return;
    float a0 = bo[0], a1 = bo[1], a2 = bo[2];
    const float* hp = h + (long)i * GNN;
#pragma unroll 4
    for (int k = 0; k < GNN; ++k) {
        float v = hp[k];
        a0 += v * Wo[k * 3 + 0];
        a1 += v * Wo[k * 3 + 1];
        a2 += v * Wo[k * 3 + 2];
    }
    out[i * 3 + 0] = a0; out[i * 3 + 1] = a1; out[i * 3 + 2] = a2;
}

// ---------------------------------------------------------------------------
extern "C" void kernel_launch(void* const* d_in, const int* in_sizes, int n_in,
                              void* d_out, int out_size, void* d_ws, size_t ws_size,
                              hipStream_t stream)
{
    const float* x    = (const float*)d_in[0];
    const int*   ei   = (const int*)d_in[1];
    const float* W1_0 = (const float*)d_in[2];
    const float* b1_0 = (const float*)d_in[3];
    const float* W2_0 = (const float*)d_in[4];
    const float* b2_0 = (const float*)d_in[5];
    const float* W1s  = (const float*)d_in[6];
    const float* b1s  = (const float*)d_in[7];
    const float* W2s  = (const float*)d_in[8];
    const float* b2s  = (const float*)d_in[9];
    const float* Wo   = (const float*)d_in[10];
    const float* bo   = (const float*)d_in[11];

    const int* src = ei;
    const int* dst = ei + N_EDGES;

    // workspace carve-up (all 16B aligned)
    float*  hA    = (float*)d_ws;
    float*  hB    = hA + (long)N_NODES * GNN;
    __bf16* w1_0p = (__bf16*)(hB + (long)N_NODES * GNN);
    __bf16* w2_0p = w1_0p + 32 * GNN;
    __bf16* w1p   = w2_0p + GNN * GNN;            // 3 x (256*128)
    __bf16* w2p   = w1p + 3L * 256 * GNN;         // 3 x (128*128)

    const long NH = (long)N_NODES * GNN;
    const int  ZB = (int)((NH + 255) / 256);

    // 1) pre-swizzle all weights to fragment layout (bf16)
    prep_weight<<<(32 * GNN + 255) / 256, 256, 0, stream>>>(W1_0, 14, 32, w1_0p);
    prep_weight<<<(GNN * GNN + 255) / 256, 256, 0, stream>>>(W2_0, GNN, GNN, w2_0p);
    for (int i = 0; i < 3; ++i) {
        prep_weight<<<(256 * GNN + 255) / 256, 256, 0, stream>>>(
            W1s + (long)i * 256 * GNN, 256, 256, w1p + (long)i * 256 * GNN);
        prep_weight<<<(GNN * GNN + 255) / 256, 256, 0, stream>>>(
            W2s + (long)i * GNN * GNN, GNN, GNN, w2p + (long)i * GNN * GNN);
    }

    const int EB = N_EDGES / EPB; // 9375, exact

    // 2) layer 0
    zero_kernel<<<ZB, 256, 0, stream>>>(hA, NH);
    edgeconv_first<<<EB, WAVES * 32, 0, stream>>>(x, src, dst,
                                                  w1_0p, b1_0, w2_0p, b2_0, hA);

    // 3) layers 1..3 (ping-pong)
    float* cur = hA;
    float* nxt = hB;
    for (int i = 0; i < 3; ++i) {
        zero_kernel<<<ZB, 256, 0, stream>>>(nxt, NH);
        edgeconv_mid<<<EB, WAVES * 32, 0, stream>>>(
            cur, src, dst,
            w1p + (long)i * 256 * GNN, b1s + (long)i * GNN,
            w2p + (long)i * GNN * GNN, b2s + (long)i * GNN,
            nxt);
        float* t = cur; cur = nxt; nxt = t;
    }

    // 4) output head
    head_kernel<<<(N_NODES + 255) / 256, 256, 0, stream>>>(
        cur, Wo, bo, (float*)d_out, N_NODES);
}
